// PyTorchTernaryKernel_43112881717530
// MI455X (gfx1250) — matrix operand order repaired and checked by reading, not compiled, not run on hardware
//
#include <hip/hip_runtime.h>

#define OUT_F  8192
#define IN_F   4096
#define TOKENS 512

typedef __attribute__((ext_vector_type(16))) _Float16 v16h;
typedef __attribute__((ext_vector_type(8)))  _Float16 v8h;
typedef __attribute__((ext_vector_type(8)))  float    v8f;

// LDS row stride in halves: 64 data + 8 pad = 72 halves = 144 bytes.
// 144 % 16 == 0 (b128-aligned rows); bank step 36 (gcd(36,64)=4 -> 16 distinct
// banks across 16 consecutive rows, conflict-free fragment reads).
#define LDS_STRIDE 72

static __device__ __forceinline__ v16h cat8(v8h lo, v8h hi) {
  return __builtin_shufflevector(lo, hi, 0, 1, 2, 3, 4, 5, 6, 7,
                                         8, 9, 10, 11, 12, 13, 14, 15);
}

__global__ __launch_bounds__(256)
void ternary_wmma_gemm(const float* __restrict__ x,
                       const int*   __restrict__ wq,
                       const float* __restrict__ alpha,
                       const float* __restrict__ bias,
                       float* __restrict__ out)
{
  __shared__ unsigned long long lut[256];          // byte -> 4 packed f16 (+1/0/-1)
  __shared__ _Float16 xs[64   * LDS_STRIDE];       // x tile,   64M x 64K f16
  __shared__ _Float16 wsd[256 * LDS_STRIDE];       // W tile,  256N x 64K f16

  const int tid  = threadIdx.x;
  const int lane = tid & 31;
  const int wid  = tid >> 5;
  const int wm   = wid >> 2;        // 0..1  (M position of wave)
  const int wn   = wid & 3;         // 0..3  (N position of wave)
  const int hi16 = lane >> 4;       // half-wave select
  const int l15  = lane & 15;

  const int bm = blockIdx.y * 64;   // token tile base
  const int bn = blockIdx.x * 256;  // output-feature tile base

  // ---- build ternary decode LUT: code 0 -> -1, 2 -> +1, 1/3 -> 0 ----
  {
    unsigned long long q = 0;
    #pragma unroll
    for (int j = 0; j < 4; ++j) {
      int c = (tid >> (2 * j)) & 3;
      unsigned long long h = (c & 1) ? 0ull : ((c == 0) ? 0xBC00ull : 0x3C00ull);
      q |= h << (16 * j);
    }
    lut[tid] = q;
  }

  const v8f vzero = {};
  v8f acc[2][4];
  #pragma unroll
  for (int i = 0; i < 2; ++i)
    #pragma unroll
    for (int j = 0; j < 4; ++j)
      acc[i][j] = vzero;

  __syncthreads();

  // per-thread staging geometry (fixed; only k0 advances)
  const int xr = tid >> 2;                  // 0..63  : x row within tile
  const int xc = (tid & 3) * 16;            // 0/16/32/48 : k column group
  const float* xsrc = x + (size_t)(bm + xr) * IN_F + xc;

  const int* wsrc = wq + (size_t)(bn + tid) * (IN_F / 4);  // one W row per thread

  for (int k0 = 0; k0 < IN_F; k0 += 64) {
    if (k0 + 64 < IN_F) {                   // pull next slab toward L2/L0
      __builtin_prefetch(xsrc + k0 + 64, 0, 3);
      __builtin_prefetch(wsrc + ((k0 + 64) >> 2), 0, 3);
    }

    // ---- stage x: 64 x 64 f32 -> f16 in LDS (b128 loads, b128 stores) ----
    {
      union F4 { float4 v; float f[4]; } f[4];
      const float4* p = (const float4*)(xsrc + k0);
      #pragma unroll
      for (int q = 0; q < 4; ++q) f[q].v = p[q];
      v8h h0, h1;
      #pragma unroll
      for (int q = 0; q < 8; ++q) {
        h0[q] = (_Float16)f[q >> 2].f[q & 3];
        h1[q] = (_Float16)f[2 + (q >> 2)].f[q & 3];
      }
      *(v8h*)(xs + xr * LDS_STRIDE + xc)     = h0;
      *(v8h*)(xs + xr * LDS_STRIDE + xc + 8) = h1;
    }

    // ---- stage W: 1 row/thread, 16 int32 words -> LUT decode -> 64 f16 ----
    {
      union I4 { int4 v; int s[4]; } wv[4];
      const int4* p = (const int4*)(wsrc + (k0 >> 2));
      #pragma unroll
      for (int q = 0; q < 4; ++q) wv[q].v = p[q];
      unsigned long long* dst = (unsigned long long*)(wsd + tid * LDS_STRIDE);
      #pragma unroll
      for (int q = 0; q < 16; ++q)
        dst[q] = lut[wv[q >> 2].s[q & 3] & 255];
    }
    __syncthreads();

    // ---- 2 K-slabs of 32, 2x4 WMMA tiles each ----
    #pragma unroll
    for (int ks = 0; ks < 2; ++ks) {
      // A layout (16-bit A 16x32): lane<16 holds K{0..7,16..23}, lane>=16 K{8..15,24..31}
      const int ka = ks * 32 + hi16 * 8;
      // B layout (32x16): lane<16 holds K{0..7,16..23}, lane>=16 K{8..15,24..31}
      const int kb = ks * 32 + hi16 * 8;

      v16h a[2], b[4];
      #pragma unroll
      for (int i = 0; i < 2; ++i) {
        const _Float16* pa = xs + (wm * 32 + i * 16 + l15) * LDS_STRIDE + ka;
        a[i] = cat8(*(const v8h*)pa, *(const v8h*)(pa + 16));
      }
      #pragma unroll
      for (int j = 0; j < 4; ++j) {
        const _Float16* pb = wsd + (wn * 64 + j * 16 + l15) * LDS_STRIDE + kb;
        b[j] = cat8(*(const v8h*)pb, *(const v8h*)(pb + 16));
      }
      #pragma unroll
      for (int i = 0; i < 2; ++i)
        #pragma unroll
        for (int j = 0; j < 4; ++j)
          acc[i][j] = __builtin_amdgcn_wmma_f32_16x16x32_f16(
              false, a[i], false, b[j], (short)0, acc[i][j], false, false);
    }
    __syncthreads();   // protect LDS before next staging pass
  }

  // ---- epilogue: out = acc * alpha[col] + bias[col] ----
  // C/D layout: VGPR e -> M = e (lane<16) or 8+e (lane>=16), N = lane&15
  #pragma unroll
  for (int j = 0; j < 4; ++j) {
    const int col = bn + wn * 64 + j * 16 + l15;
    const float al = alpha[col];
    const float bi = bias[col];
    #pragma unroll
    for (int i = 0; i < 2; ++i) {
      const int rbase = bm + wm * 32 + i * 16 + hi16 * 8;
      #pragma unroll
      for (int e = 0; e < 8; ++e)
        out[(size_t)(rbase + e) * OUT_F + col] = acc[i][j][e] * al + bi;
    }
  }
}

extern "C" void kernel_launch(void* const* d_in, const int* in_sizes, int n_in,
                              void* d_out, int out_size, void* d_ws, size_t ws_size,
                              hipStream_t stream) {
  (void)in_sizes; (void)n_in; (void)out_size; (void)d_ws; (void)ws_size;
  const float* x     = (const float*)d_in[0];
  const int*   wq    = (const int*)d_in[1];
  const float* alpha = (const float*)d_in[2];
  const float* bias  = (const float*)d_in[3];
  float*       out   = (float*)d_out;

  dim3 grid(OUT_F / 256, TOKENS / 64);   // 32 x 8 blocks
  dim3 block(256);                        // 8 wave32s
  ternary_wmma_gemm<<<grid, block, 0, stream>>>(x, wq, alpha, bias, out);
}
